// GCN_46067819216956
// MI455X (gfx1250) — matrix operand order repaired and checked by reading, not compiled, not run on hardware
//
#include <hip/hip_runtime.h>

#define NNODES 100000
#define FDIM 128
#define LROW 136  // padded LDS row stride in shorts (272B) to avoid bank conflicts

typedef __attribute__((ext_vector_type(16))) __bf16 v16bf;
typedef __attribute__((ext_vector_type(16))) unsigned short v16u16;
typedef __attribute__((ext_vector_type(8)))  unsigned short v8u16;
typedef __attribute__((ext_vector_type(8)))  float v8f;

union V16U { v16u16 v; v8u16 h[2]; };

static __device__ __forceinline__ unsigned short f2bf(float f) {
  // round-to-nearest-even fp32 -> bf16
  unsigned int u = __float_as_uint(f);
  u += 0x7FFFu + ((u >> 16) & 1u);
  return (unsigned short)(u >> 16);
}

static __device__ __forceinline__ void atomAddF32(float* p, float v) {
  __hip_atomic_fetch_add(p, v, __ATOMIC_RELAXED, __HIP_MEMORY_SCOPE_AGENT);
}

// ---------------- degree / dinv ----------------
__global__ void __launch_bounds__(256) k_fill1(float* p, int n) {
  int i = blockIdx.x * 256 + threadIdx.x;
  if (i < n) p[i] = 1.0f;  // self-loop included in degree
}

__global__ void __launch_bounds__(256) k_degcount(const int* __restrict__ dst,
                                                  float* __restrict__ deg, int E) {
  int stride = gridDim.x * 256;
  for (int i = blockIdx.x * 256 + threadIdx.x; i < E; i += stride)
    atomAddF32(&deg[dst[i]], 1.0f);
}

__global__ void __launch_bounds__(256) k_rsqrt(float* p, int n) {
  int i = blockIdx.x * 256 + threadIdx.x;
  if (i < n) p[i] = rsqrtf(p[i]);
}

// ---------------- WMMA GEMM: Out[M,128] = X[M,128] @ W[128,128] ----------------
// 256 threads = 8 waves; block computes a 128x128 tile, wave owns a 16x128 strip.
// Both X-tile and W staged in LDS as bf16 (X row-major, W transposed), converted
// once during coalesced staging; inner loop is pure ds_load_b128 + v_wmma.
__global__ void __launch_bounds__(256) k_gemm_wmma(const float* __restrict__ X,
                                                   const float* __restrict__ W,
                                                   float* __restrict__ Out, int M) {
  __shared__ unsigned short ldsX[128 * LROW];  // X tile [r][k] bf16, padded rows
  __shared__ unsigned short ldsW[128 * LROW];  // Wt   [n][k] bf16, padded rows

  const int rowBase0 = blockIdx.x * 128;

  // stage X tile: coalesced float4 reads, convert, vector LDS store (zero-pad OOB rows)
  const float4* X4 = (const float4*)X;
  for (int i4 = threadIdx.x; i4 < 128 * 32; i4 += 256) {
    int row = i4 >> 5, c4 = i4 & 31;
    float4 v = make_float4(0.f, 0.f, 0.f, 0.f);
    if (rowBase0 + row < M) v = X4[(size_t)(rowBase0 + row) * 32 + c4];
    ushort4 s;
    s.x = f2bf(v.x); s.y = f2bf(v.y); s.z = f2bf(v.z); s.w = f2bf(v.w);
    *(ushort4*)&ldsX[row * LROW + c4 * 4] = s;
  }
  // stage W transposed: coalesced float4 reads, scalar transposed stores (once per block)
  for (int i4 = threadIdx.x; i4 < 128 * 32; i4 += 256) {
    int k = i4 >> 5, n0 = (i4 & 31) * 4;
    float4 v = ((const float4*)W)[i4];
    ldsW[(n0 + 0) * LROW + k] = f2bf(v.x);
    ldsW[(n0 + 1) * LROW + k] = f2bf(v.y);
    ldsW[(n0 + 2) * LROW + k] = f2bf(v.z);
    ldsW[(n0 + 3) * LROW + k] = f2bf(v.w);
  }
  __syncthreads();

  const int lane = threadIdx.x & 31;
  const int wv   = threadIdx.x >> 5;
  const int rowBase = rowBase0 + wv * 16;
  if (rowBase >= M) return;  // M % 16 == 0: tiles all-in or all-out; no barriers follow

  const int rl = lane & 15;       // row within tile (A) / col (B,C)
  const int g  = lane >> 4;       // half-wave group
  const int lrow = wv * 16 + rl;  // LDS row of this lane's A rows

  v8f acc[8] = {};
#pragma unroll
  for (int ks = 0; ks < 4; ++ks) {
    // A fragment (16-bit A 16x32 layout): K runs [32ks+8g,+8) and [32ks+16+8g,+8)
    V16U a;
    a.h[0] = *(const v8u16*)&ldsX[lrow * LROW + ks * 32 + g * 8];
    a.h[1] = *(const v8u16*)&ldsX[lrow * LROW + ks * 32 + 16 + g * 8];
    v16bf av = __builtin_bit_cast(v16bf, a.v);
#pragma unroll
    for (int nt = 0; nt < 8; ++nt) {
      // B fragment: col n = nt*16+rl; K runs [32ks+8g,+8) and [32ks+16+8g,+8)
      V16U b;
      const unsigned short* bp = &ldsW[(nt * 16 + rl) * LROW + ks * 32 + g * 8];
      b.h[0] = *(const v8u16*)bp;
      b.h[1] = *(const v8u16*)(bp + 16);
      v16bf bv = __builtin_bit_cast(v16bf, b.v);
      acc[nt] = __builtin_amdgcn_wmma_f32_16x16x32_bf16(
          false, av, false, bv, (short)0, acc[nt], false, false);
    }
  }

  // C/D layout: VGPR i, lane: row = rowBase + 8*g + i, col = nt*16 + rl
#pragma unroll
  for (int nt = 0; nt < 8; ++nt) {
    float* obase = Out + (size_t)(rowBase + g * 8) * FDIM + nt * 16 + rl;
#pragma unroll
    for (int i = 0; i < 8; ++i) obase[(size_t)i * FDIM] = acc[nt][i];
  }
}

// ---------------- zero / aggregate / finalize ----------------
__global__ void __launch_bounds__(256) k_zero4(float4* p, int n4) {
  int i = blockIdx.x * 256 + threadIdx.x;
  if (i < n4) p[i] = make_float4(0.f, 0.f, 0.f, 0.f);
}

// one wave per edge: lane handles 4 features; scatter-add via relaxed agent-scope
// f32 atomics (native global_atomic_add_f32, L2-resident working set)
__global__ void __launch_bounds__(256) k_aggregate(const float* __restrict__ h,
                                                   const int* __restrict__ src,
                                                   const int* __restrict__ dst,
                                                   const float* __restrict__ dinv,
                                                   float* __restrict__ agg, int E) {
  int lane = threadIdx.x & 31;
  int wid  = (blockIdx.x * 256 + threadIdx.x) >> 5;
  int nw   = (gridDim.x * 256) >> 5;
  for (int e = wid; e < E; e += nw) {
    int s = src[e], d = dst[e];
    float norm = dinv[s] * dinv[d];
    float4 v = ((const float4*)(h + (size_t)s * FDIM))[lane];
    float* ap = agg + (size_t)d * FDIM + lane * 4;
    atomAddF32(ap + 0, v.x * norm);
    atomAddF32(ap + 1, v.y * norm);
    atomAddF32(ap + 2, v.z * norm);
    atomAddF32(ap + 3, v.w * norm);
  }
}

// agg := relu(agg + h * dinv^2 + bias)   (in place over agg)
__global__ void __launch_bounds__(256) k_finalize(float* __restrict__ agg,
                                                  const float* __restrict__ h,
                                                  const float* __restrict__ dinv,
                                                  const float* __restrict__ bias,
                                                  int total) {
  int i = blockIdx.x * 256 + threadIdx.x;
  if (i >= total) return;
  int node = i >> 7, f = i & 127;
  float di = dinv[node];
  float v = agg[i] + h[i] * di * di + bias[f];
  agg[i] = fmaxf(v, 0.0f);
}

// out[node] = dot(h[node,:], w) + b ; one wave per node, shuffle reduction
__global__ void __launch_bounds__(256) k_linear(const float* __restrict__ h,
                                                const float* __restrict__ w,
                                                const float* __restrict__ b,
                                                float* __restrict__ out, int N) {
  int lane = threadIdx.x & 31;
  int node = (blockIdx.x * 256 + threadIdx.x) >> 5;
  if (node >= N) return;
  float4 hv = ((const float4*)(h + (size_t)node * FDIM))[lane];
  float4 wv = ((const float4*)w)[lane];
  float s = hv.x * wv.x + hv.y * wv.y + hv.z * wv.z + hv.w * wv.w;
#pragma unroll
  for (int off = 16; off > 0; off >>= 1) s += __shfl_down(s, off, 32);
  if (lane == 0) out[node] = s + b[0];
}

extern "C" void kernel_launch(void* const* d_in, const int* in_sizes, int n_in,
                              void* d_out, int out_size, void* d_ws, size_t ws_size,
                              hipStream_t stream) {
  (void)n_in; (void)out_size; (void)ws_size;
  const float* x    = (const float*)d_in[0];
  const int*   ei   = (const int*)d_in[1];
  const float* W1   = (const float*)d_in[2];
  const float* b1   = (const float*)d_in[3];
  const float* W2   = (const float*)d_in[4];
  const float* b2   = (const float*)d_in[5];
  const float* Wlin = (const float*)d_in[6];
  const float* blin = (const float*)d_in[7];
  float* out = (float*)d_out;

  const int N = NNODES;
  const int E = in_sizes[1] / 2;
  const int* src = ei;
  const int* dst = ei + E;

  float* ws   = (float*)d_ws;
  float* dinv = ws;                              // N floats
  float* bufA = ws + N;                          // N*128 floats
  float* bufB = bufA + (size_t)N * FDIM;         // N*128 floats

  const int total = N * FDIM;                    // 12.8M
  const int gemmBlocks = (N + 127) / 128;        // 782
  const int nBlk = (N + 255) / 256;
  const int eBlk = 2048;

  // degree -> dinv
  k_fill1<<<nBlk, 256, 0, stream>>>(dinv, N);
  k_degcount<<<eBlk, 256, 0, stream>>>(dst, dinv, E);
  k_rsqrt<<<nBlk, 256, 0, stream>>>(dinv, N);

  // ---- layer 1 ----
  k_gemm_wmma<<<gemmBlocks, 256, 0, stream>>>(x, W1, bufA, N);
  k_zero4<<<(total / 4 + 255) / 256, 256, 0, stream>>>((float4*)bufB, total / 4);
  k_aggregate<<<4096, 256, 0, stream>>>(bufA, src, dst, dinv, bufB, E);
  k_finalize<<<(total + 255) / 256, 256, 0, stream>>>(bufB, bufA, dinv, b1, total);

  // ---- layer 2 ---- (bufB = h1_relu consumed by GEMM, then reused for agg)
  k_gemm_wmma<<<gemmBlocks, 256, 0, stream>>>(bufB, W2, bufA, N);
  k_zero4<<<(total / 4 + 255) / 256, 256, 0, stream>>>((float4*)bufB, total / 4);
  k_aggregate<<<4096, 256, 0, stream>>>(bufA, src, dst, dinv, bufB, E);
  k_finalize<<<(total + 255) / 256, 256, 0, stream>>>(bufB, bufA, dinv, b2, total);

  // ---- final linear ----
  k_linear<<<(N * 32 + 255) / 256, 256, 0, stream>>>(bufB, Wlin, blin, out, N);
}
